// TextureDiffusionIterative_62603443306756
// MI455X (gfx1250) — matrix operand order repaired and checked by reading, not compiled, not run on hardware
//
#include <hip/hip_runtime.h>
#include <hip/hip_fp16.h>

typedef __attribute__((ext_vector_type(16))) _Float16 v16h;
typedef __attribute__((ext_vector_type(8)))  float    v8f;

constexpr int Bn = 2, Hh = 256, Ww = 256, Cc = 24, KK = 49;
constexpr int HW   = Hh * Ww;          // 65536
constexpr int MCH  = Cc * KK;          // 1176 predictor output channels
constexpr int MT   = (MCH + 15) / 16;  // 74 M-tiles of 16
constexpr int MCHP = MT * 16;          // 1184 padded rows in kernel tensor
constexpr int KIN  = 27;               // 3 channels * 3x3 taps, padded to 32 for WMMA

// ---------------------------------------------------------------------------
// Repack w_kp (1176 x 27, row-major) into WMMA A-fragment layout, fp16.
// Per ISA 7.12.2 (16-bit A 16x32): lane l holds row M = 16t + (l&15);
// halves h: K = h + (h>=8 ? 8 : 0) + (l>=16 ? 8 : 0).  K>=27 and M>=1176 -> 0.
// ---------------------------------------------------------------------------
__global__ void repack_wkp(const float* __restrict__ wkp, _Float16* __restrict__ apack) {
    int t = blockIdx.x;            // 0..73
    int l = threadIdx.x;           // 0..31
    int m = t * 16 + (l & 15);
    int kbase = (l >> 4) << 3;     // 0 or 8
    _Float16* dst = apack + ((size_t)(t * 32 + l)) * 16;
    #pragma unroll
    for (int h = 0; h < 16; ++h) {
        int k = h + ((h >> 3) << 3) + kbase;
        float v = 0.f;
        if (m < MCH && k < KIN) v = wkp[m * KIN + k];
        dst[h] = (_Float16)v;
    }
}

// ---------------------------------------------------------------------------
// depth_proj: 3x3 conv, 1 -> 24 channels, SAME padding.
// ---------------------------------------------------------------------------
__global__ void __launch_bounds__(256)
depth_proj(const float* __restrict__ depth, const float* __restrict__ wdp,
           const float* __restrict__ bdp, float* __restrict__ latent) {
    __shared__ float sw[Cc * 9];
    __shared__ float sb[Cc];
    int tid = threadIdx.x;
    for (int i = tid; i < Cc * 9; i += 256) sw[i] = wdp[i];
    if (tid < Cc) sb[tid] = bdp[tid];
    __syncthreads();

    int gp = blockIdx.x * 256 + tid;     // 0 .. B*HW-1
    int b = gp / HW, p = gp % HW;
    int y = p >> 8, x = p & 255;

    float patch[9];
    #pragma unroll
    for (int dy = -1; dy <= 1; ++dy)
        #pragma unroll
        for (int dx = -1; dx <= 1; ++dx) {
            int yy = y + dy, xx = x + dx;
            patch[(dy + 1) * 3 + dx + 1] =
                (yy >= 0 && yy < Hh && xx >= 0 && xx < Ww) ? depth[b * HW + (yy << 8) + xx] : 0.f;
        }
    #pragma unroll
    for (int c = 0; c < Cc; ++c) {
        float acc = sb[c];
        #pragma unroll
        for (int i = 0; i < 9; ++i) acc += sw[c * 9 + i] * patch[i];
        latent[((size_t)b * Cc + c) * HW + p] = acc;
    }
}

// ---------------------------------------------------------------------------
// kernel_predictor conv (WMMA GEMM) fused with channel softmax statistics.
// Workgroup = 128 threads (4 waves) handles 32 consecutive pixels (two
// N-tiles sharing each A fragment -> 2 WMMAs per A load, 64B-aligned stores).
// Pass 1: WMMA sweep, running MAX only (branchless cndmask for padded rows).
// Pass 2: WMMA sweep, e = exp(logit - max) stored UNNORMALIZED in fp16;
//         per-pixel 1/sum written to a small invs array; normalization is
//         folded into the diffusion kernel (diffused = inv[p] * sum e*x).
// ---------------------------------------------------------------------------
__global__ void __launch_bounds__(128)
kp_wmma_softmax(const float* __restrict__ tex, const _Float16* __restrict__ apack,
                const float* __restrict__ bkp, _Float16* __restrict__ kernels,
                float* __restrict__ invs) {
    __shared__ _Float16 sB[32 * 32];      // im2col B operand, [K][N], 32 pixels
    __shared__ float sBias[MCHP];         // 1184 biases (padded rows = 0)
    __shared__ float sM0[128], sM1[128];  // cross-wave max reduction
    __shared__ float sS0[128], sS1[128];  // cross-wave sum reduction

    int tid  = threadIdx.x;
    int lane = tid & 31;
    int wave = tid >> 5;

    int wg   = blockIdx.x;                // 0 .. B*HW/32-1
    int bimg = wg >> 11;                  // / (HW/32)
    int p0   = (wg & 2047) << 5;          // first pixel (row-aligned, 32-wide)
    int y    = p0 >> 8, x0 = p0 & 255;

    for (int i = tid; i < MCHP; i += 128)
        sBias[i] = (i < MCH) ? bkp[i] : 0.f;

    // Build im2col B (32 pixels, K padded 27->32 with zeros)
    for (int i = tid; i < 32 * 32; i += 128) {
        int k = i >> 5, n32 = i & 31;
        float v = 0.f;
        if (k < KIN) {
            int ch = k / 9, tap = k % 9;
            int dy = tap / 3 - 1, dx = tap % 3 - 1;
            int yy = y + dy, xx = x0 + n32 + dx;
            if (yy >= 0 && yy < Hh && xx >= 0 && xx < Ww)
                v = tex[((size_t)(bimg * 3 + ch) * Hh + yy) * Ww + xx];
        }
        sB[i] = (_Float16)v;
    }
    __syncthreads();

    // B fragments: lane holds column N = lane&15; K = (h&7) + (h>=8 ? 16 : 0) + (lane>=16 ? 8 : 0)
    int n  = lane & 15;
    int kb = (lane >> 4) << 3;
    v16h bfrag0, bfrag1;
    #pragma unroll
    for (int h = 0; h < 16; ++h) {
        bfrag0[h] = sB[(kb + (h & 7) + ((h >> 3) << 4)) * 32 + n];
        bfrag1[h] = sB[(kb + (h & 7) + ((h >> 3) << 4)) * 32 + n + 16];
    }

    int hi = (lane >> 4) << 3;            // C/D row offset: lanes>=16 hold M = r+8
    float m0 = -3.0e38f, m1 = -3.0e38f;

    // ---- pass 1: logits -> per-lane running max (no exp, no branches) ----
    for (int t = wave; t < MT; t += 4) {
        v16h afrag = *(const v16h*)(apack + ((size_t)(t * 32 + lane)) * 16);
        v8f acc0, acc1;
        #pragma unroll
        for (int r = 0; r < 8; ++r) { float bz = sBias[t * 16 + hi + r]; acc0[r] = bz; acc1[r] = bz; }
        acc0 = __builtin_amdgcn_wmma_f32_16x16x32_f16(false, afrag, false, bfrag0,
                                                      (short)0, acc0, false, false);
        acc1 = __builtin_amdgcn_wmma_f32_16x16x32_f16(false, afrag, false, bfrag1,
                                                      (short)0, acc1, false, false);
        #pragma unroll
        for (int r = 0; r < 8; ++r) {
            bool valid = (t * 16 + hi + r) < MCH;
            m0 = fmaxf(m0, valid ? acc0[r] : -1.0e30f);
            m1 = fmaxf(m1, valid ? acc1[r] : -1.0e30f);
        }
    }

    // ---- reduce max across the 8 lane-groups that share each pixel ----
    sM0[tid] = m0; sM1[tid] = m1;
    __syncthreads();
    float Mf0 = -3.0e38f, Mf1 = -3.0e38f;
    #pragma unroll
    for (int w = 0; w < 8; ++w) {
        int src = (w >> 1) * 32 + ((w & 1) << 4) + n;   // 4 waves x 2 half-waves
        Mf0 = fmaxf(Mf0, sM0[src]);
        Mf1 = fmaxf(Mf1, sM1[src]);
    }

    // ---- pass 2: recompute logits, emit unnormalized fp16 exp, sum ----
    float s0 = 0.f, s1 = 0.f;
    for (int t = wave; t < MT; t += 4) {
        v16h afrag = *(const v16h*)(apack + ((size_t)(t * 32 + lane)) * 16);
        v8f acc0, acc1;
        #pragma unroll
        for (int r = 0; r < 8; ++r) { float bz = sBias[t * 16 + hi + r]; acc0[r] = bz; acc1[r] = bz; }
        acc0 = __builtin_amdgcn_wmma_f32_16x16x32_f16(false, afrag, false, bfrag0,
                                                      (short)0, acc0, false, false);
        acc1 = __builtin_amdgcn_wmma_f32_16x16x32_f16(false, afrag, false, bfrag1,
                                                      (short)0, acc1, false, false);
        #pragma unroll
        for (int r = 0; r < 8; ++r) {
            int mm = t * 16 + hi + r;
            bool valid = mm < MCH;
            float e0 = __expf(acc0[r] - Mf0);
            float e1 = __expf(acc1[r] - Mf1);
            s0 += valid ? e0 : 0.f;
            s1 += valid ? e1 : 0.f;
            // rows >= MCH land in the padded region, never read by diffuse
            size_t base = ((size_t)bimg * MCHP + mm) * HW + p0 + n;
            kernels[base]      = (_Float16)e0;
            kernels[base + 16] = (_Float16)e1;
        }
    }

    // ---- reduce sums, write per-pixel 1/sum ----
    sS0[tid] = s0; sS1[tid] = s1;
    __syncthreads();
    if (tid < 32) {
        int nn = tid & 15;
        float S = 0.f;
        #pragma unroll
        for (int w = 0; w < 8; ++w) {
            int src = (w >> 1) * 32 + ((w & 1) << 4) + nn;
            S += (tid < 16) ? sS0[src] : sS1[src];
        }
        invs[(size_t)bimg * HW + p0 + tid] = 1.f / S;
    }
}

// ---------------------------------------------------------------------------
// One diffusion iteration:
//   dst[b,c,p] = invs[b,p] * sum_kk e[b,c*49+kk,p] * src_halo
// Block = 16x16 pixel tile for one (b,c); src tile (22x22, halo 3) in LDS.
// ---------------------------------------------------------------------------
__global__ void __launch_bounds__(256)
diffuse(const float* __restrict__ src, const _Float16* __restrict__ kernels,
        const float* __restrict__ invs, float* __restrict__ dst) {
    __shared__ float tile[22 * 22];
    int tx = threadIdx.x & 15, ty = threadIdx.x >> 4;
    int x0 = blockIdx.x << 4, y0 = blockIdx.y << 4;
    int bc = blockIdx.z;                          // b*24 + c
    int b = bc / Cc, c = bc % Cc;

    const float* sp = src + (size_t)bc * HW;
    for (int i = threadIdx.x; i < 22 * 22; i += 256) {
        int ly = i / 22, lx = i % 22;
        int gy = y0 + ly - 3, gx = x0 + lx - 3;
        tile[i] = (gy >= 0 && gy < Hh && gx >= 0 && gx < Ww) ? sp[(gy << 8) + gx] : 0.f;
    }
    __syncthreads();

    int p = ((y0 + ty) << 8) + x0 + tx;
    const _Float16* kp = kernels + ((size_t)b * MCHP + c * KK) * HW + p;
    float acc = 0.f;
    #pragma unroll
    for (int kh = 0; kh < 7; ++kh)
        #pragma unroll
        for (int kw = 0; kw < 7; ++kw) {
            float wv = (float)kp[(size_t)(kh * 7 + kw) * HW];
            acc += wv * tile[(ty + kh) * 22 + tx + kw];
        }
    dst[(size_t)bc * HW + p] = acc * invs[(size_t)b * HW + p];
}

// ---------------------------------------------------------------------------
// to_depth: 1x1 conv 24 -> 1.
// ---------------------------------------------------------------------------
__global__ void __launch_bounds__(256)
to_depth_k(const float* __restrict__ diff, const float* __restrict__ wtd,
           const float* __restrict__ btd, float* __restrict__ out) {
    __shared__ float sw[Cc];
    __shared__ float sb0;
    if (threadIdx.x < Cc) sw[threadIdx.x] = wtd[threadIdx.x];
    if (threadIdx.x == 0) sb0 = btd[0];
    __syncthreads();
    int gp = blockIdx.x * 256 + threadIdx.x;
    int b = gp / HW, p = gp % HW;
    float acc = sb0;
    #pragma unroll
    for (int c = 0; c < Cc; ++c) acc += sw[c] * diff[((size_t)b * Cc + c) * HW + p];
    out[gp] = acc;
}

extern "C" void kernel_launch(void* const* d_in, const int* in_sizes, int n_in,
                              void* d_out, int out_size, void* d_ws, size_t ws_size,
                              hipStream_t stream) {
    (void)in_sizes; (void)n_in; (void)out_size; (void)ws_size;
    const float* depth   = (const float*)d_in[0];
    const float* texture = (const float*)d_in[1];
    const float* w_dp    = (const float*)d_in[2];
    const float* b_dp    = (const float*)d_in[3];
    const float* w_kp    = (const float*)d_in[4];
    const float* b_kp    = (const float*)d_in[5];
    const float* w_td    = (const float*)d_in[6];
    const float* b_td    = (const float*)d_in[7];
    float* out = (float*)d_out;

    char* ws = (char*)d_ws;
    size_t off = 0;
    _Float16* apack = (_Float16*)(ws + off); off += ((size_t)MT * 32 * 16 * 2 + 255) & ~(size_t)255;
    float*    latent = (float*)(ws + off);   off += (size_t)Bn * Cc * HW * 4;
    float*    ping   = (float*)(ws + off);   off += (size_t)Bn * Cc * HW * 4;
    float*    pong   = (float*)(ws + off);   off += (size_t)Bn * Cc * HW * 4;
    float*    invs   = (float*)(ws + off);   off += (size_t)Bn * HW * 4;
    _Float16* kern   = (_Float16*)(ws + off);   // B * 1184 * HW fp16 (padded rows)

    repack_wkp<<<MT, 32, 0, stream>>>(w_kp, apack);
    depth_proj<<<Bn * HW / 256, 256, 0, stream>>>(depth, w_dp, b_dp, latent);
    kp_wmma_softmax<<<Bn * HW / 32, 128, 0, stream>>>(texture, apack, b_kp, kern, invs);

    dim3 dgrid(Ww / 16, Hh / 16, Bn * Cc);
    diffuse<<<dgrid, 256, 0, stream>>>(latent, kern, invs, ping);   // iter 0
    diffuse<<<dgrid, 256, 0, stream>>>(ping,   kern, invs, pong);   // iter 1
    diffuse<<<dgrid, 256, 0, stream>>>(pong,   kern, invs, ping);   // iter 2
    diffuse<<<dgrid, 256, 0, stream>>>(ping,   kern, invs, pong);   // iter 3

    to_depth_k<<<Bn * HW / 256, 256, 0, stream>>>(pong, w_td, b_td, out);
}
